// LSTM_7524782702916
// MI455X (gfx1250) — compile-verified, matrix operand order checked by reading, not run on hardware
//
#include <hip/hip_runtime.h>
#include <hip/hip_bf16.h>

// ---------- types ----------
typedef __attribute__((ext_vector_type(16))) _Float16 v16h;
typedef __attribute__((ext_vector_type(8)))  _Float16 v8h;
typedef __attribute__((ext_vector_type(8)))  float    v8f;

#define WMMA_F32_F16(a, b, c) \
  __builtin_amdgcn_wmma_f32_16x16x32_f16(false, (a), false, (b), (short)0, (c), false, false)

// Load one A/B fragment from a row-major f16 row (global or LDS pointer).
// Caller passes k already including the lane-dependent koff (0 or 8).
// Chunks: [k, k+8) and [k+16, k+24). 16-byte aligned when k % 8 == 0.
template <typename PtrT>
__device__ __forceinline__ v16h load_row_f16(PtrT row, int k) {
  v8h lo = *(const v8h*)(row + k);
  v8h hi = *(const v8h*)(row + k + 16);
  v16h r;
#pragma unroll
  for (int i = 0; i < 8; ++i) { r[i] = lo[i]; r[i + 8] = hi[i]; }
  return r;
}

__device__ __forceinline__ v16h load_row_f32(const float* __restrict__ row, int k) {
  v8f lo = *(const v8f*)(row + k);
  v8f hi = *(const v8f*)(row + k + 16);
  v16h r;
#pragma unroll
  for (int i = 0; i < 8; ++i) { r[i] = (_Float16)lo[i]; r[i + 8] = (_Float16)hi[i]; }
  return r;
}

__device__ __forceinline__ float sigmoidf_(float x) {
  return 1.0f / (1.0f + __expf(-x));
}

// ---------- elementwise helpers ----------
__global__ void f32_to_f16_kernel(const float* __restrict__ src, _Float16* __restrict__ dst, int n) {
  int i = blockIdx.x * blockDim.x + threadIdx.x;
  if (i < n) dst[i] = (_Float16)src[i];
}

__global__ void init_state_kernel(float* __restrict__ c0, float* __restrict__ c1,
                                  _Float16* __restrict__ h, int n) {
  int i = blockIdx.x * blockDim.x + threadIdx.x;
  if (i < n) { c0[i] = 0.0f; c1[i] = 0.0f; h[i] = (_Float16)0.0f; }
}

// ---------- Phase A0: Zpre = x @ Wih0^T + bih0 + bhh0 ----------
// A: x f32 [B,S,F] viewed as [M=32768, K=128] with m = s*64 + b
// W: f16 [N=2048, K=128] row-major.  D: f32 [32768, 2048] ([s,b,j]).
__global__ void gemm_x_wih_kernel(const float* __restrict__ x, const _Float16* __restrict__ W,
                                  const float* __restrict__ b1, const float* __restrict__ b2,
                                  float* __restrict__ D) {
  const int tid = threadIdx.x, wave = tid >> 5, lane = tid & 31;
  const int t  = blockIdx.x * 8 + wave;          // 2048 m-tiles x 128 n-tiles
  const int mt = t >> 7, nt = t & 127;
  const int l15 = lane & 15;
  const int koff = (lane & 16) ? 8 : 0;
  const int m  = mt * 16 + l15;
  const int bb = m & 63, ss = m >> 6;
  const float*    arow = x + ((size_t)bb * 512 + ss) * 128;   // x[b, s, :]
  const _Float16* wrow = W + (size_t)(nt * 16 + l15) * 128;
  v8f acc = {};
#pragma unroll
  for (int k0 = 0; k0 < 128; k0 += 32) {
    v16h af = load_row_f32(arow, k0 + koff);
    v16h bf = load_row_f16(wrow, k0 + koff);
    acc = WMMA_F32_F16(af, bf, acc);
  }
  const int ncol = nt * 16 + l15;
  const float bias = b1[ncol] + b2[ncol];
  const int row0 = mt * 16 + ((lane & 16) ? 8 : 0);
#pragma unroll
  for (int r = 0; r < 8; ++r)
    D[(size_t)(row0 + r) * 2048 + ncol] = acc[r] + bias;
}

// ---------- fused recurrent step ----------
// Launch s (0..512) runs, in ONE kernel:
//   blocks  0..15 : layer-0 step s      (if s < 512):  z0 = zpre0[s] + h0[s-1] @ Whh0^T
//   blocks 16..31 : layer-1 step t=s-1  (if s >= 1) :  z1 = bias1 + hs0[t] @ Wih1^T + h1[t-1] @ Whh1^T
// Each block owns 32 hidden units (all four gates). Operand panels (64x512 f16) are staged
// into LDS with a 520-half row stride (conflict-free ds_load_b128); the gate-exchange
// buffer aliases the same LDS after the K-loop.
#define HSTRIDE 520
#define PANEL_BYTES (64 * HSTRIDE * 2)   // 66560
__global__ void lstm_fused_step_kernel(const int s,
                                       const float* __restrict__ zpre0,   // [512][64,2048]
                                       const _Float16* __restrict__ Whh0, // f16 [2048,512]
                                       _Float16* __restrict__ hs0,        // f16 [512][64,512]
                                       const _Float16* __restrict__ Wih1, // f16 [2048,512]
                                       const _Float16* __restrict__ Whh1, // f16 [2048,512]
                                       const float* __restrict__ bih1,
                                       const float* __restrict__ bhh1,
                                       _Float16* __restrict__ hs1,        // f16 [512][64,512]
                                       float* __restrict__ c0,
                                       float* __restrict__ c1,
                                       const _Float16* __restrict__ hzero) {
  extern __shared__ char smem[];                  // 2 * PANEL_BYTES
  _Float16* hshA = (_Float16*)smem;               // panel A (padded rows)
  _Float16* hshB = (_Float16*)(smem + PANEL_BYTES); // panel B (layer-1 only)
  float*    zsh  = (float*)smem;                  // aliases panel A AFTER the K-loop (32 KB)

  const int lay = blockIdx.x >> 4;
  if (lay == 0) { if (s >= 512) return; }
  else          { if (s == 0)   return; }
  const int t = s - 1;                            // layer-1 time index
  const int nbase = (blockIdx.x & 15) * 32;

  const int tid = threadIdx.x, wave = tid >> 5, lane = tid & 31;
  const int gate = wave >> 1, nt = wave & 1;
  const int l15 = lane & 15;
  const int koff = (lane & 16) ? 8 : 0;
  const int hi8  = (lane & 16) ? 8 : 0;

  // ---- stage operand panel(s): 64 rows x 512 halfs = 4096 b128 chunks each ----
  const _Float16* srcA = (lay == 0)
      ? ((s == 0) ? hzero : hs0 + (size_t)(s - 1) * 32768)   // h0[s-1]
      : hs0 + (size_t)t * 32768;                             // hs0[t]
#pragma unroll
  for (int i = 0; i < 16; ++i) {
    const int chunk = tid + i * 256;
    const int row = chunk >> 6;
    const int col = (chunk & 63) * 8;
    *(v8h*)(hshA + row * HSTRIDE + col) = *(const v8h*)(srcA + row * 512 + col);
  }
  if (lay == 1) {
    const _Float16* srcB = (t == 0) ? hzero : hs1 + (size_t)(t - 1) * 32768;  // h1[t-1]
#pragma unroll
    for (int i = 0; i < 16; ++i) {
      const int chunk = tid + i * 256;
      const int row = chunk >> 6;
      const int col = (chunk & 63) * 8;
      *(v8h*)(hshB + row * HSTRIDE + col) = *(const v8h*)(srcB + row * 512 + col);
    }
  } else {
    // prefetch this block's zpre gating footprint (64 b x 4 gates x 128B lines)
    // so the L2 fetch overlaps the K-loop.  (lowers to global_prefetch_b8)
    const float* zp = zpre0 + (size_t)s * 131072;
    const int pb = tid >> 2, pg = tid & 3;
    __builtin_prefetch(zp + (size_t)pb * 2048 + pg * 512 + nbase, 0, 1);
  }
  __syncthreads();

  const int j = gate * 512 + nbase + nt * 16 + l15;    // weight row feeding this B-column
  const _Float16* wrowA = ((lay == 0) ? Whh0 : Wih1) + (size_t)j * 512;

  v8f acc0 = {}, acc1 = {}, acc2 = {}, acc3 = {};
#pragma unroll 2
  for (int k0 = 0; k0 < 512; k0 += 32) {
    const int k = k0 + koff;
    v16h bf = load_row_f16(wrowA, k);
    v16h a0 = load_row_f16(hshA + (l15)      * HSTRIDE, k);
    v16h a1 = load_row_f16(hshA + (16 + l15) * HSTRIDE, k);
    v16h a2 = load_row_f16(hshA + (32 + l15) * HSTRIDE, k);
    v16h a3 = load_row_f16(hshA + (48 + l15) * HSTRIDE, k);
    acc0 = WMMA_F32_F16(a0, bf, acc0);
    acc1 = WMMA_F32_F16(a1, bf, acc1);
    acc2 = WMMA_F32_F16(a2, bf, acc2);
    acc3 = WMMA_F32_F16(a3, bf, acc3);
  }
  if (lay == 1) {
    const _Float16* wrowB = Whh1 + (size_t)j * 512;
#pragma unroll 2
    for (int k0 = 0; k0 < 512; k0 += 32) {
      const int k = k0 + koff;
      v16h bf = load_row_f16(wrowB, k);
      v16h a0 = load_row_f16(hshB + (l15)      * HSTRIDE, k);
      v16h a1 = load_row_f16(hshB + (16 + l15) * HSTRIDE, k);
      v16h a2 = load_row_f16(hshB + (32 + l15) * HSTRIDE, k);
      v16h a3 = load_row_f16(hshB + (48 + l15) * HSTRIDE, k);
      acc0 = WMMA_F32_F16(a0, bf, acc0);
      acc1 = WMMA_F32_F16(a1, bf, acc1);
      acc2 = WMMA_F32_F16(a2, bf, acc2);
      acc3 = WMMA_F32_F16(a3, bf, acc3);
    }
  }
  __syncthreads();   // everyone done reading panels; safe to reuse LDS as zsh

  const int cidx = gate * 32 + nt * 16 + l15;
#pragma unroll
  for (int r = 0; r < 8; ++r) {
    zsh[(hi8 + r)      * 128 + cidx] = acc0[r];
    zsh[(16 + hi8 + r) * 128 + cidx] = acc1[r];
    zsh[(32 + hi8 + r) * 128 + cidx] = acc2[r];
    zsh[(48 + hi8 + r) * 128 + cidx] = acc3[r];
  }
  __syncthreads();

  float*     cc = (lay == 0) ? c0 : c1;
  _Float16*  ho = (lay == 0) ? hs0 + (size_t)s * 32768 : hs1 + (size_t)t * 32768;
  const float* zp = zpre0 + (size_t)s * 131072;

#pragma unroll
  for (int e = 0; e < 8; ++e) {
    const int idx = e * 256 + tid;      // 2048 (b, nl) pairs
    const int b = idx >> 5, nl = idx & 31;
    const int ng = nbase + nl;
    float bi, bf_, bg, bo;
    if (lay == 0) {
      const size_t zrow = (size_t)b * 2048;
      bi  = zp[zrow + ng];
      bf_ = zp[zrow + 512 + ng];
      bg  = zp[zrow + 1024 + ng];
      bo  = zp[zrow + 1536 + ng];
    } else {
      bi  = bih1[ng]        + bhh1[ng];
      bf_ = bih1[512 + ng]  + bhh1[512 + ng];
      bg  = bih1[1024 + ng] + bhh1[1024 + ng];
      bo  = bih1[1536 + ng] + bhh1[1536 + ng];
    }
    const float zi = zsh[b * 128 + nl]      + bi;
    const float zf = zsh[b * 128 + 32 + nl] + bf_;
    const float zg = zsh[b * 128 + 64 + nl] + bg;
    const float zo = zsh[b * 128 + 96 + nl] + bo;
    const float ig = sigmoidf_(zi);
    const float fg = sigmoidf_(zf);
    const float gg = tanhf(zg);
    const float og = sigmoidf_(zo);
    const int ci = b * 512 + ng;
    const float cv = fg * cc[ci] + ig * gg;
    cc[ci] = cv;
    ho[ci] = (_Float16)(og * tanhf(cv));
  }
}

// ---------- final linear: out[b,o] = hs1_flat[b,:] @ Wlin[o,:] + blin[o] ----------
// hs1: f16 [s, b, n] -> A[b, k] with k = s*512 + n.  Wlin: f16 [256, 262144].
__global__ void final_linear_kernel(const _Float16* __restrict__ hs1,
                                    const _Float16* __restrict__ Wlin,
                                    const float* __restrict__ blin,
                                    float* __restrict__ out) {
  const int tid = threadIdx.x, wave = tid >> 5, lane = tid & 31;
  const int t  = blockIdx.x * 8 + wave;           // 4 m-tiles x 16 n-tiles
  const int mt = t >> 4, nt = t & 15;
  const int l15 = lane & 15;
  const int koff = (lane & 16) ? 8 : 0;
  const int b = mt * 16 + l15;
  const _Float16* wrow = Wlin + (size_t)(nt * 16 + l15) * 262144;
  v8f acc = {};
  for (int k0 = 0; k0 < 262144; k0 += 32) {
    // whole 32-chunk stays within one timestep s = k0 >> 9 (512 % 32 == 0)
    const _Float16* arow = hs1 + (size_t)(k0 >> 9) * 32768 + (size_t)b * 512;
    v16h af = load_row_f16(arow, (k0 & 511) + koff);
    v16h bf = load_row_f16(wrow, k0 + koff);
    acc = WMMA_F32_F16(af, bf, acc);
  }
  const int ncol = nt * 16 + l15;
  const float bias = blin[ncol];
  const int row0 = mt * 16 + ((lane & 16) ? 8 : 0);
#pragma unroll
  for (int r = 0; r < 8; ++r)
    out[(size_t)(row0 + r) * 256 + ncol] = acc[r] + bias;
}

// ---------- host ----------
extern "C" void kernel_launch(void* const* d_in, const int* in_sizes, int n_in,
                              void* d_out, int out_size, void* d_ws, size_t ws_size,
                              hipStream_t stream) {
  (void)in_sizes; (void)n_in; (void)out_size; (void)ws_size;
  const float* x    = (const float*)d_in[0];
  const float* Wih0 = (const float*)d_in[1];
  const float* Whh0 = (const float*)d_in[2];
  const float* bih0 = (const float*)d_in[3];
  const float* bhh0 = (const float*)d_in[4];
  const float* Wih1 = (const float*)d_in[5];
  const float* Whh1 = (const float*)d_in[6];
  const float* bih1 = (const float*)d_in[7];
  const float* bhh1 = (const float*)d_in[8];
  const float* Wlin = (const float*)d_in[9];
  const float* blin = (const float*)d_in[10];
  float* out = (float*)d_out;

  // scratch carve-out
  size_t off = 0;
  char* base = (char*)d_ws;
  auto carve = [&](size_t bytes) -> void* {
    void* p = base + off;
    off += (bytes + 255) & ~(size_t)255;
    return p;
  };
  float*     zpre  = (float*)    carve((size_t)32768 * 2048 * 4);  // 256 MB
  _Float16*  hs0h  = (_Float16*) carve((size_t)32768 * 512 * 2);   // 32 MB
  _Float16*  hs1h  = (_Float16*) carve((size_t)32768 * 512 * 2);   // 32 MB
  _Float16*  Wih0h = (_Float16*) carve((size_t)2048 * 128 * 2);
  _Float16*  Whh0h = (_Float16*) carve((size_t)2048 * 512 * 2);
  _Float16*  Wih1h = (_Float16*) carve((size_t)2048 * 512 * 2);
  _Float16*  Whh1h = (_Float16*) carve((size_t)2048 * 512 * 2);
  _Float16*  Wlinh = (_Float16*) carve((size_t)256 * 262144 * 2);  // 128 MB
  _Float16*  hzero = (_Float16*) carve((size_t)64 * 512 * 2);      // h_{-1} == 0
  float*     c0buf = (float*)    carve((size_t)64 * 512 * 4);
  float*     c1buf = (float*)    carve((size_t)64 * 512 * 4);

  const dim3 blk(256);
  const int step_lds = 2 * PANEL_BYTES;   // 133120 bytes

  // weight conversions (f32 -> f16)
  f32_to_f16_kernel<<<(2048 * 128 + 255) / 256, blk, 0, stream>>>(Wih0, Wih0h, 2048 * 128);
  f32_to_f16_kernel<<<(2048 * 512 + 255) / 256, blk, 0, stream>>>(Whh0, Whh0h, 2048 * 512);
  f32_to_f16_kernel<<<(2048 * 512 + 255) / 256, blk, 0, stream>>>(Wih1, Wih1h, 2048 * 512);
  f32_to_f16_kernel<<<(2048 * 512 + 255) / 256, blk, 0, stream>>>(Whh1, Whh1h, 2048 * 512);
  f32_to_f16_kernel<<<(67108864 + 255) / 256, blk, 0, stream>>>(Wlin, Wlinh, 67108864);

  // input-projection GEMM for layer 0 (all timesteps at once)
  gemm_x_wih_kernel<<<32768, blk, 0, stream>>>(x, Wih0h, bih0, bhh0, zpre);
  init_state_kernel<<<128, blk, 0, stream>>>(c0buf, c1buf, hzero, 32768);

  // fused recurrence: launch s runs layer0 step s and layer1 step s-1
  for (int s = 0; s <= 512; ++s) {
    lstm_fused_step_kernel<<<32, blk, step_lds, stream>>>(
        s, zpre, Whh0h, hs0h, Wih1h, Whh1h, bih1, bhh1, hs1h, c0buf, c1buf, hzero);
  }

  // final linear
  final_linear_kernel<<<8, blk, 0, stream>>>(hs1h, Wlinh, blin, out);
}
